// MapleSparseMoeBlock_49074296324447
// MI455X (gfx1250) — compile-verified, matrix operand order checked by reading, not run on hardware
//
#include <hip/hip_runtime.h>
#include <hip/hip_bf16.h>

#define HID   2048
#define ITR   1408
#define NEXP  16
#define NTOK  4096   // BATCH * SEQ

typedef __attribute__((ext_vector_type(16))) __bf16 v16bf;
typedef __attribute__((ext_vector_type(2)))  __bf16 bf16x2;
typedef __attribute__((ext_vector_type(8)))  float  v8f;
typedef __attribute__((ext_vector_type(2)))  float  f32x2;

__device__ __forceinline__ float fast_silu(float g) {
    // g * sigmoid(g) with v_rcp_f32 instead of IEEE divide
    return g * __builtin_amdgcn_rcpf(1.0f + __expf(-g));
}

__device__ __forceinline__ void load_frag_f32(v16bf& d, const float* p, int k0, int khalf) {
#pragma unroll
    for (int v = 0; v < 8; ++v) {
        const int kk = k0 + ((v & 3) * 2) + (khalf * 8) + ((v >> 2) * 16);
        f32x2 t = *(const f32x2*)(p + kk);
        d[2 * v] = (__bf16)t.x;  d[2 * v + 1] = (__bf16)t.y;
    }
}

__device__ __forceinline__ void load_frag_bf16(v16bf& d, const __bf16* p, int k0, int khalf) {
#pragma unroll
    for (int v = 0; v < 8; ++v) {
        const int kk = k0 + ((v & 3) * 2) + (khalf * 8) + ((v >> 2) * 16);
        bf16x2 t = *(const bf16x2*)(p + kk);
        d[2 * v] = t.x;  d[2 * v + 1] = t.y;
    }
}

// ---------------------------------------------------------------- zero
__global__ __launch_bounds__(256)
void moe_zero(float* __restrict__ y, int* __restrict__ counts) {
    size_t i = (size_t)blockIdx.x * 256 + threadIdx.x;
    if (i < (size_t)NTOK * HID) y[i] = 0.f;
    if (blockIdx.x == 0 && threadIdx.x < NEXP) counts[threadIdx.x] = 0;
}

// ---------------------------------------------------------------- gate
// one wave32 per token: fp32 logits, softmax, top-2, renorm, scatter to lists
__global__ __launch_bounds__(256)
void moe_gate(const float* __restrict__ x, const float* __restrict__ gw,
              int* __restrict__ counts, int* __restrict__ tok_id,
              float* __restrict__ tok_w)
{
    const int wave = threadIdx.x >> 5;
    const int lane = threadIdx.x & 31;
    const int t = blockIdx.x * 8 + wave;
    if (t >= NTOK) return;

    const float* xr = x + (size_t)t * HID;
    float xv[HID / 32];
#pragma unroll
    for (int j = 0; j < HID / 32; ++j) xv[j] = xr[lane + j * 32];

    float logit[NEXP];
#pragma unroll
    for (int e = 0; e < NEXP; ++e) {
        const float* g = gw + (size_t)e * HID;
        float acc = 0.f;
#pragma unroll
        for (int j = 0; j < HID / 32; ++j) acc = fmaf(xv[j], g[lane + j * 32], acc);
#pragma unroll
        for (int off = 16; off > 0; off >>= 1) acc += __shfl_xor(acc, off, 32);
        logit[e] = acc;   // all lanes hold the full logit vector
    }

    if (lane == 0) {
        float m = logit[0];
#pragma unroll
        for (int e = 1; e < NEXP; ++e) m = fmaxf(m, logit[e]);
        float s = 0.f, p1v = -1.f, p2v = -1.f;
        int   p1i = 0, p2i = 0;
#pragma unroll
        for (int e = 0; e < NEXP; ++e) {
            float pe = __expf(logit[e] - m);
            s += pe;
            if (pe > p1v)      { p2v = p1v; p2i = p1i; p1v = pe; p1i = e; }
            else if (pe > p2v) { p2v = pe;  p2i = e; }
        }
        float pr0 = p1v / s, pr1 = p2v / s;
        float inv = 1.f / (pr0 + pr1 + 1e-20f);
        float w0 = pr0 * inv, w1 = pr1 * inv;

        int s0 = atomicAdd(&counts[p1i], 1);
        tok_id[p1i * NTOK + s0] = t;  tok_w[p1i * NTOK + s0] = w0;
        int s1 = atomicAdd(&counts[p2i], 1);
        tok_id[p2i * NTOK + s1] = t;  tok_w[p2i * NTOK + s1] = w1;
    }
}

// ---------------------------------------------------------------- prefix
__global__ void moe_prefix(const int* __restrict__ counts, int* __restrict__ offs) {
    if (threadIdx.x == 0 && blockIdx.x == 0) {
        int a = 0;
#pragma unroll
        for (int e = 0; e < NEXP; ++e) { offs[e] = a; a += counts[e]; }
    }
}

// ---------------------------------------------------------------- gate/up GEMM + SiLU
// grid: (ITR/128, NTOK/32, NEXP); 8 waves; each wave: 32 tokens x 16 cols,
// 4 WMMAs per K-step against 4 loaded fragments (A0,A1 shared across Bg,Bu)
__global__ __launch_bounds__(256)
void moe_gu(const float* __restrict__ x, const float* __restrict__ wg,
            const float* __restrict__ wu, const int* __restrict__ counts,
            const int* __restrict__ offs, const int* __restrict__ tok_id,
            __bf16* __restrict__ h_buf)
{
    const int e   = blockIdx.z;
    const int cnt = counts[e];
    const int m0  = blockIdx.y * 32;
    if (m0 >= cnt) return;

    const int wave  = threadIdx.x >> 5;
    const int lane  = threadIdx.x & 31;
    const int khalf = lane >> 4;
    const int l16   = lane & 15;
    const int n     = blockIdx.x * 128 + wave * 16 + l16;   // inter-dim column

    int s0 = m0 + l16;      if (s0 >= cnt) s0 = cnt - 1;    // clamp tail rows
    int s1 = m0 + 16 + l16; if (s1 >= cnt) s1 = cnt - 1;
    const float* ap0 = x + (size_t)tok_id[e * NTOK + s0] * HID;
    const float* ap1 = x + (size_t)tok_id[e * NTOK + s1] * HID;
    const float* bgp = wg + ((size_t)e * ITR + n) * HID;
    const float* bup = wu + ((size_t)e * ITR + n) * HID;

    v8f cg0 = {}, cg1 = {}, cu0 = {}, cu1 = {};
    for (int k0 = 0; k0 < HID; k0 += 32) {
        v16bf a0, a1, bg, bu;
        load_frag_f32(a0, ap0, k0, khalf);
        load_frag_f32(a1, ap1, k0, khalf);
        load_frag_f32(bg, bgp, k0, khalf);
        load_frag_f32(bu, bup, k0, khalf);
        cg0 = __builtin_amdgcn_wmma_f32_16x16x32_bf16(false, a0, false, bg, (short)0, cg0, false, false);
        cu0 = __builtin_amdgcn_wmma_f32_16x16x32_bf16(false, a0, false, bu, (short)0, cu0, false, false);
        cg1 = __builtin_amdgcn_wmma_f32_16x16x32_bf16(false, a1, false, bg, (short)0, cg1, false, false);
        cu1 = __builtin_amdgcn_wmma_f32_16x16x32_bf16(false, a1, false, bu, (short)0, cu1, false, false);
    }

    const int hbase = offs[e] + m0;
#pragma unroll
    for (int r = 0; r < 8; ++r) {
        const int M = r + khalf * 8;
        if (m0 + M < cnt)
            h_buf[(size_t)(hbase + M) * ITR + n] = (__bf16)(fast_silu(cg0[r]) * cu0[r]);
        if (m0 + 16 + M < cnt)
            h_buf[(size_t)(hbase + 16 + M) * ITR + n] = (__bf16)(fast_silu(cg1[r]) * cu1[r]);
    }
}

// ---------------------------------------------------------------- down GEMM + weighted combine
// grid: (HID/256, NTOK/32, NEXP); 8 waves; each wave: 32 rows x 32 cols,
// 4 WMMAs per K-step against 4 loaded fragments (A0,A1 x B0,B1)
__global__ __launch_bounds__(256)
void moe_down(const __bf16* __restrict__ h_buf, const float* __restrict__ wd,
              const int* __restrict__ counts, const int* __restrict__ offs,
              const int* __restrict__ tok_id, const float* __restrict__ tok_w,
              float* __restrict__ y)
{
    const int e   = blockIdx.z;
    const int cnt = counts[e];
    const int m0  = blockIdx.y * 32;
    if (m0 >= cnt) return;

    const int wave  = threadIdx.x >> 5;
    const int lane  = threadIdx.x & 31;
    const int khalf = lane >> 4;
    const int l16   = lane & 15;
    const int nbase = blockIdx.x * 256 + wave * 32;
    const int n0    = nbase + l16;          // hidden-dim columns
    const int n1    = nbase + 16 + l16;

    int s0 = m0 + l16;      if (s0 >= cnt) s0 = cnt - 1;
    int s1 = m0 + 16 + l16; if (s1 >= cnt) s1 = cnt - 1;
    const __bf16* ap0 = h_buf + (size_t)(offs[e] + s0) * ITR;
    const __bf16* ap1 = h_buf + (size_t)(offs[e] + s1) * ITR;
    const float*  bp0 = wd + ((size_t)e * HID + n0) * ITR;
    const float*  bp1 = wd + ((size_t)e * HID + n1) * ITR;

    v8f c00 = {}, c01 = {}, c10 = {}, c11 = {};
    for (int k0 = 0; k0 < ITR; k0 += 32) {
        v16bf a0, a1, b0, b1;
        load_frag_bf16(a0, ap0, k0, khalf);
        load_frag_bf16(a1, ap1, k0, khalf);
        load_frag_f32 (b0, bp0, k0, khalf);
        load_frag_f32 (b1, bp1, k0, khalf);
        c00 = __builtin_amdgcn_wmma_f32_16x16x32_bf16(false, a0, false, b0, (short)0, c00, false, false);
        c01 = __builtin_amdgcn_wmma_f32_16x16x32_bf16(false, a0, false, b1, (short)0, c01, false, false);
        c10 = __builtin_amdgcn_wmma_f32_16x16x32_bf16(false, a1, false, b0, (short)0, c10, false, false);
        c11 = __builtin_amdgcn_wmma_f32_16x16x32_bf16(false, a1, false, b1, (short)0, c11, false, false);
    }

#pragma unroll
    for (int r = 0; r < 8; ++r) {
        const int M = r + khalf * 8;
        const int r0 = m0 + M, r1 = m0 + 16 + M;
        if (r0 < cnt) {
            const int   t   = tok_id[e * NTOK + r0];
            const float wgt = tok_w[e * NTOK + r0];
            atomicAdd(&y[(size_t)t * HID + n0], wgt * c00[r]);
            atomicAdd(&y[(size_t)t * HID + n1], wgt * c01[r]);
        }
        if (r1 < cnt) {
            const int   t   = tok_id[e * NTOK + r1];
            const float wgt = tok_w[e * NTOK + r1];
            atomicAdd(&y[(size_t)t * HID + n0], wgt * c10[r]);
            atomicAdd(&y[(size_t)t * HID + n1], wgt * c11[r]);
        }
    }
}

// ---------------------------------------------------------------- launcher
extern "C" void kernel_launch(void* const* d_in, const int* in_sizes, int n_in,
                              void* d_out, int out_size, void* d_ws, size_t ws_size,
                              hipStream_t stream) {
    (void)in_sizes; (void)n_in; (void)out_size; (void)ws_size;
    const float* x  = (const float*)d_in[0];   // [T, H]
    const float* gw = (const float*)d_in[1];   // [E, H]
    const float* wg = (const float*)d_in[2];   // [E, I, H]
    const float* wu = (const float*)d_in[3];   // [E, I, H]
    const float* wd = (const float*)d_in[4];   // [E, H, I]
    float* y = (float*)d_out;                  // [T, H]

    // workspace layout
    int*    counts = (int*)d_ws;                                   // 16
    int*    offs   = counts + 16;                                  // 16
    int*    tok_id = (int*)((char*)d_ws + 256);                    // [E, T]
    float*  tok_w  = (float*)(tok_id + NEXP * NTOK);               // [E, T]
    __bf16* h_buf  = (__bf16*)((char*)d_ws + (1u << 20));          // [T*2, ITR] bf16 (~22 MB)

    moe_zero<<<((size_t)NTOK * HID + 255) / 256, 256, 0, stream>>>(y, counts);
    moe_gate<<<NTOK / 8, 256, 0, stream>>>(x, gw, counts, tok_id, tok_w);
    moe_prefix<<<1, 32, 0, stream>>>(counts, offs);
    moe_gu  <<<dim3(ITR / 128, NTOK / 32, NEXP), 256, 0, stream>>>(x, wg, wu, counts, offs, tok_id, h_buf);
    moe_down<<<dim3(HID / 256, NTOK / 32, NEXP), 256, 0, stream>>>(h_buf, wd, counts, offs, tok_id, tok_w, y);
}
